// Head_22393959481557
// MI455X (gfx1250) — matrix operand order repaired and checked by reading, not compiled, not run on hardware
//
#include <hip/hip_runtime.h>

#define B_ 8
#define T_ 2048
#define E_ 1024
#define D_ 64

typedef _Float16 h2  __attribute__((ext_vector_type(2)));
typedef _Float16 h4  __attribute__((ext_vector_type(4)));
typedef _Float16 v16h __attribute__((ext_vector_type(16)));
typedef float    v8f  __attribute__((ext_vector_type(8)));

// ---------------------------------------------------------------------------
// Fragment loaders matching CDNA5 WMMA VGPR layouts (ISA 7.12.2, wave32).
// ---------------------------------------------------------------------------

// A-matrix 16x32 f16, row-major source with row stride ld (elements).
// Lane l holds row m = l&15; VGPR v holds K = 16*(v>>2) + 8*(l>>4) + 2*(v&3), +1.
__device__ __forceinline__ v16h load_a_frag(const _Float16* __restrict__ src,
                                            int ld, int lane) {
    v16h a;
    const int m  = lane & 15;
    const int hi = (lane >> 4) & 1;
    const _Float16* row = src + (size_t)m * ld;
#pragma unroll
    for (int v = 0; v < 8; ++v) {
        const int k0 = 16 * (v >> 2) + 8 * hi + 2 * (v & 3);
        h2 t = *(const h2*)(row + k0);
        a[2 * v]     = t[0];
        a[2 * v + 1] = t[1];
    }
    return a;
}

// B-matrix 32x16 f16 (KxN). Source stored n-major: element(k,n) = src[n*ld + k].
// Lane l holds col n = l&15; VGPR v holds K = 16*(v>>2) + 8*(l>>4) + 2*(v&3), +1.
__device__ __forceinline__ v16h load_b_frag(const _Float16* __restrict__ src,
                                            int ld, int lane) {
    v16h b;
    const int n     = lane & 15;
    const int kbase = (lane >> 4) ? 8 : 0;
    const _Float16* col = src + (size_t)n * ld + kbase;
#pragma unroll
    for (int v = 0; v < 8; ++v) {
        h2 t = *(const h2*)(col + 16 * (v >> 2) + 2 * (v & 3));
        b[2 * v]     = t[0];
        b[2 * v + 1] = t[1];
    }
    return b;
}

__device__ __forceinline__ v8f wmma_f16(v16h a, v16h b, v8f c) {
    return __builtin_amdgcn_wmma_f32_16x16x32_f16(
        /*neg_a=*/false, a, /*neg_b=*/false, b,
        /*c_mod=*/(short)0, c, /*reuse_a=*/false, /*reuse_b=*/false);
}

// ---------------------------------------------------------------------------
// Kernel 1: x f32 -> f16 (vectorized 4 elems/thread)
// ---------------------------------------------------------------------------
__global__ void cvt_x_kernel(const float* __restrict__ in,
                             _Float16* __restrict__ out, int n4) {
    int i = blockIdx.x * blockDim.x + threadIdx.x;
    if (i >= n4) return;
    float4 v = ((const float4*)in)[i];
    h4 o;
    o[0] = (_Float16)v.x; o[1] = (_Float16)v.y;
    o[2] = (_Float16)v.z; o[3] = (_Float16)v.w;
    ((h4*)out)[i] = o;
}

// ---------------------------------------------------------------------------
// Kernel 2: W [E,D] f32 -> Wt [D,E] f16 (d-major so B-fragment pairs are
// contiguous along E)
// ---------------------------------------------------------------------------
__global__ void transpose_w_kernel(const float* __restrict__ W,
                                   _Float16* __restrict__ Wt) {
    int tid = blockIdx.x * blockDim.x + threadIdx.x;   // flat over D_*E_
    int d = tid >> 10;          // / E_
    int e = tid & (E_ - 1);     // % E_
    Wt[tid] = (_Float16)W[(size_t)e * D_ + d];
}

// ---------------------------------------------------------------------------
// Kernel 3: projections. One block = 16 t-rows; wave w owns d-tile w (16 cols)
// and produces Q, K (row-major f16) and Vt (d-major f16) via WMMA over E.
// ---------------------------------------------------------------------------
__global__ __launch_bounds__(128) void proj_kernel(
    const _Float16* __restrict__ xh,
    const _Float16* __restrict__ Wtq,
    const _Float16* __restrict__ Wtk,
    const _Float16* __restrict__ Wtv,
    _Float16* __restrict__ Qh,
    _Float16* __restrict__ Kh,
    _Float16* __restrict__ Vth) {
    const int lane = threadIdx.x & 31;
    const int w    = threadIdx.x >> 5;              // d-tile 0..3
    const int b    = blockIdx.x / (T_ / 16);
    const int t0   = (blockIdx.x % (T_ / 16)) * 16;

    const _Float16* xrow = xh + (size_t)(b * T_ + t0) * E_;
    const _Float16* wq = Wtq + (size_t)(w * 16) * E_;
    const _Float16* wk = Wtk + (size_t)(w * 16) * E_;
    const _Float16* wv = Wtv + (size_t)(w * 16) * E_;

    v8f cq = {}; v8f ck = {}; v8f cv = {};
#pragma unroll 4
    for (int e0 = 0; e0 < E_; e0 += 32) {
        v16h a = load_a_frag(xrow + e0, E_, lane);
        cq = wmma_f16(a, load_b_frag(wq + e0, E_, lane), cq);
        ck = wmma_f16(a, load_b_frag(wk + e0, E_, lane), ck);
        cv = wmma_f16(a, load_b_frag(wv + e0, E_, lane), cv);
    }

    const int n = lane & 15, hi = lane >> 4;
#pragma unroll
    for (int r = 0; r < 8; ++r) {
        const int m = r + 8 * hi;
        const size_t qk_idx = (size_t)(b * T_ + t0 + m) * D_ + w * 16 + n;
        Qh[qk_idx] = (_Float16)cq[r];
        Kh[qk_idx] = (_Float16)ck[r];
        Vth[((size_t)b * D_ + w * 16 + n) * T_ + t0 + m] = (_Float16)cv[r];
    }
}

// ---------------------------------------------------------------------------
// Kernel 4: flash attention. 4 waves/block, wave owns 16 query rows.
// Key tiles of 32; S via 4 WMMA, online softmax, P->LDS->A-frag, PV via 4 WMMA.
// ---------------------------------------------------------------------------
#define PSTR 40   // LDS row stride (halves) for the 16x32 P tile

__global__ __launch_bounds__(128) void attn_kernel(
    const _Float16* __restrict__ Qh,
    const _Float16* __restrict__ Kh,
    const _Float16* __restrict__ Vth,
    float* __restrict__ out) {
    __shared__ _Float16 pbuf[4][16][PSTR];

    const int lane = threadIdx.x & 31;
    const int w    = threadIdx.x >> 5;
    const int b    = blockIdx.x / (T_ / 64);
    const int q0   = (blockIdx.x % (T_ / 64)) * 64 + w * 16;

    const _Float16* Qb  = Qh  + (size_t)b * T_ * D_;
    const _Float16* Kb  = Kh  + (size_t)b * T_ * D_;
    const _Float16* Vtb = Vth + (size_t)b * D_ * T_;

    const v16h aq0 = load_a_frag(Qb + (size_t)q0 * D_,      D_, lane);
    const v16h aq1 = load_a_frag(Qb + (size_t)q0 * D_ + 32, D_, lane);

    v8f acc[4];
    float mx[8], ls[8];
#pragma unroll
    for (int t = 0; t < 4; ++t) acc[t] = (v8f){};
#pragma unroll
    for (int r = 0; r < 8; ++r) { mx[r] = -3.0e38f; ls[r] = 0.0f; }

    const int n = lane & 15, hi = lane >> 4;
    _Float16* ptile = &pbuf[w][0][0];

    for (int j0 = 0; j0 < q0 + 16; j0 += 32) {
        // --- S = Q * K^T (scaled), 16 queries x 32 keys ---
        const _Float16* kp0 = Kb + (size_t)j0 * D_;
        const _Float16* kp1 = Kb + (size_t)(j0 + 16) * D_;
        v8f s0 = {}; v8f s1 = {};
        s0 = wmma_f16(aq0, load_b_frag(kp0,      D_, lane), s0);
        s0 = wmma_f16(aq1, load_b_frag(kp0 + 32, D_, lane), s0);
        s1 = wmma_f16(aq0, load_b_frag(kp1,      D_, lane), s1);
        s1 = wmma_f16(aq1, load_b_frag(kp1 + 32, D_, lane), s1);

        const bool need_mask = (j0 + 31 > q0);
#pragma unroll
        for (int r = 0; r < 8; ++r) {
            s0[r] *= 0.125f;            // D=64 -> 1/sqrt(64)
            s1[r] *= 0.125f;
            if (need_mask) {
                const int q = q0 + r + 8 * hi;
                if (j0 + n      > q) s0[r] = -1.0e30f;
                if (j0 + 16 + n > q) s1[r] = -1.0e30f;
            }
        }

        // --- online softmax (row = 16 lanes of a half-wave) ---
        float p0[8], p1[8], alpha[8];
#pragma unroll
        for (int r = 0; r < 8; ++r) {
            float rm = fmaxf(s0[r], s1[r]);
            rm = fmaxf(rm, __shfl_xor(rm, 1, 32));
            rm = fmaxf(rm, __shfl_xor(rm, 2, 32));
            rm = fmaxf(rm, __shfl_xor(rm, 4, 32));
            rm = fmaxf(rm, __shfl_xor(rm, 8, 32));
            const float mnew = fmaxf(mx[r], rm);
            const float al   = __expf(mx[r] - mnew);
            const float e0v  = __expf(s0[r] - mnew);
            const float e1v  = __expf(s1[r] - mnew);
            float rs = e0v + e1v;
            rs += __shfl_xor(rs, 1, 32);
            rs += __shfl_xor(rs, 2, 32);
            rs += __shfl_xor(rs, 4, 32);
            rs += __shfl_xor(rs, 8, 32);
            ls[r] = ls[r] * al + rs;
            mx[r] = mnew;
            alpha[r] = al;
            p0[r] = e0v;
            p1[r] = e1v;
        }
#pragma unroll
        for (int t = 0; t < 4; ++t)
#pragma unroll
            for (int r = 0; r < 8; ++r) acc[t][r] *= alpha[r];

        // --- C-layout -> A-layout transpose of P through this wave's LDS tile ---
#pragma unroll
        for (int r = 0; r < 8; ++r) {
            const int m = r + 8 * hi;
            pbuf[w][m][n]      = (_Float16)p0[r];
            pbuf[w][m][16 + n] = (_Float16)p1[r];
        }
        asm volatile("s_wait_dscnt 0x0" ::: "memory");
        const v16h ap = load_a_frag(ptile, PSTR, lane);

        // --- Y += P * V  (Vt is d-major: pairs contiguous along keys) ---
#pragma unroll
        for (int t = 0; t < 4; ++t) {
            v16h bv = load_b_frag(Vtb + (size_t)(t * 16) * T_ + j0, T_, lane);
            acc[t] = wmma_f16(ap, bv, acc[t]);
        }
    }

    // --- epilogue: normalize and store f32 ---
    float* ob = out + (size_t)(b * T_ + q0) * D_;
#pragma unroll
    for (int r = 0; r < 8; ++r) {
        const int m = r + 8 * hi;
        const float inv = 1.0f / ls[r];
#pragma unroll
        for (int t = 0; t < 4; ++t)
            ob[(size_t)m * D_ + t * 16 + n] = acc[t][r] * inv;
    }
}

// ---------------------------------------------------------------------------
// Host launcher
// ---------------------------------------------------------------------------
extern "C" void kernel_launch(void* const* d_in, const int* in_sizes, int n_in,
                              void* d_out, int out_size, void* d_ws, size_t ws_size,
                              hipStream_t stream) {
    (void)in_sizes; (void)n_in; (void)out_size; (void)ws_size;
    const float* x  = (const float*)d_in[0];
    const float* Wk = (const float*)d_in[1];
    const float* Wq = (const float*)d_in[2];
    const float* Wv = (const float*)d_in[3];
    float* out = (float*)d_out;

    char* ws = (char*)d_ws;
    size_t off = 0;
    _Float16* xh  = (_Float16*)(ws + off); off += (size_t)B_ * T_ * E_ * 2;
    _Float16* Wtq = (_Float16*)(ws + off); off += (size_t)D_ * E_ * 2;
    _Float16* Wtk = (_Float16*)(ws + off); off += (size_t)D_ * E_ * 2;
    _Float16* Wtv = (_Float16*)(ws + off); off += (size_t)D_ * E_ * 2;
    _Float16* Qh  = (_Float16*)(ws + off); off += (size_t)B_ * T_ * D_ * 2;
    _Float16* Kh  = (_Float16*)(ws + off); off += (size_t)B_ * T_ * D_ * 2;
    _Float16* Vth = (_Float16*)(ws + off); off += (size_t)B_ * T_ * D_ * 2;

    const int n4 = B_ * T_ * E_ / 4;
    cvt_x_kernel<<<n4 / 256, 256, 0, stream>>>(x, xh, n4);
    transpose_w_kernel<<<(D_ * E_) / 256, 256, 0, stream>>>(Wq, Wtq);
    transpose_w_kernel<<<(D_ * E_) / 256, 256, 0, stream>>>(Wk, Wtk);
    transpose_w_kernel<<<(D_ * E_) / 256, 256, 0, stream>>>(Wv, Wtv);

    proj_kernel<<<B_ * T_ / 16, 128, 0, stream>>>(xh, Wtq, Wtk, Wtv, Qh, Kh, Vth);
    attn_kernel<<<B_ * (T_ / 64), 128, 0, stream>>>(Qh, Kh, Vth, out);
}
